// MultiHeadedAttentionwithEulerFormer_24661702214160
// MI455X (gfx1250) — matrix operand order repaired and checked by reading, not compiled, not run on hardware
//
#include <hip/hip_runtime.h>
#include <hip/hip_bf16.h>
#include <math.h>

// Problem constants (match reference)
#define B_     2
#define S_     2048
#define HID_   512
#define HEADS_ 8
#define DH_    64
#define MROWS  (B_ * S_)          // 4096
#define QT_MAX (S_ / 16 - 1)      // 127
#define LN10000 9.210340371976184f

typedef __attribute__((ext_vector_type(16))) _Float16 v16h;
typedef __attribute__((ext_vector_type(8)))  _Float16 v8h;
typedef __attribute__((ext_vector_type(8)))  float    v8f;

// ---------------------------------------------------------------------------
// WMMA fragment loaders (layouts per CDNA5 ISA §7.12.2, wave32).
// All loads are explicit 16B vector loads -> 2x b128 per fragment.
// ---------------------------------------------------------------------------
__device__ __forceinline__ v16h cat8(v8h a, v8h b) {
  return __builtin_shufflevector(a, b, 0, 1, 2, 3, 4, 5, 6, 7,
                                 8, 9, 10, 11, 12, 13, 14, 15);
}

// A-matrix 16x32 f16 (quad pattern): lane holds row (lane&15); its 16 values
// are two contiguous 16B runs at k = k0+(lane>>4)*8 and +16.
__device__ __forceinline__ v16h load_A_frag(const _Float16* __restrict__ base,
                                            int ld, int k0) {
  const int lane = threadIdx.x & 31;
  const _Float16* p = base + (size_t)(lane & 15) * ld + k0 + (lane >> 4) * 8;
  const v8h lo = *(const v8h*)(p);
  const v8h hi = *(const v8h*)(p + 16);
  return cat8(lo, hi);
}

// B-matrix 32x16 f16 (quad pattern) where logical
// B[k][n] = M[(n0+n)*ld + (k0+k)]: lane holds col (n0+lane&15); its 16 values
// are two contiguous 16B runs at k = k0+(lane>>4)*8 and +16.
__device__ __forceinline__ v16h load_BT_frag(const _Float16* __restrict__ M,
                                             int ld, int k0, int n0) {
  const int lane = threadIdx.x & 31;
  const _Float16* p =
      M + (size_t)(n0 + (lane & 15)) * ld + k0 + (lane >> 4) * 8;
  const v8h lo = *(const v8h*)(p);
  const v8h hi = *(const v8h*)(p + 16);
  return cat8(lo, hi);
}

// ---------------------------------------------------------------------------
// 16-lane-row max butterfly on VALU DPP (stays inside each half-wave; this is
// exactly the C-layout row grouping). Fallback: __shfl_xor (ds_bpermute).
// ---------------------------------------------------------------------------
#if __has_builtin(__builtin_amdgcn_update_dpp)
template <int CTRL>
__device__ __forceinline__ float dpp_maxstep(float x) {
  const int xi = __builtin_bit_cast(int, x);
  const int r  = __builtin_amdgcn_update_dpp(xi, xi, CTRL, 0xf, 0xf, true);
  return fmaxf(x, __builtin_bit_cast(float, r));
}
__device__ __forceinline__ float rowmax16(float x) {
  x = dpp_maxstep<0xB1>(x);   // quad_perm [1,0,3,2]  (^1)
  x = dpp_maxstep<0x4E>(x);   // quad_perm [2,3,0,1]  (^2)
  x = dpp_maxstep<0x141>(x);  // row_half_mirror      (^7)
  x = dpp_maxstep<0x140>(x);  // row_mirror           (^15)
  return x;
}
#else
__device__ __forceinline__ float rowmax16(float x) {
  x = fmaxf(x, __shfl_xor(x, 1));
  x = fmaxf(x, __shfl_xor(x, 2));
  x = fmaxf(x, __shfl_xor(x, 4));
  x = fmaxf(x, __shfl_xor(x, 8));
  return x;
}
#endif

// ---------------------------------------------------------------------------
// f32 -> f16 convert
// ---------------------------------------------------------------------------
__global__ __launch_bounds__(256) void f32_to_f16_kernel(
    const float* __restrict__ in, _Float16* __restrict__ out, int n) {
  int i = blockIdx.x * blockDim.x + threadIdx.x;
  if (i < n) out[i] = (_Float16)in[i];
}

// ---------------------------------------------------------------------------
// GEMM: Y = X @ W^T + bias.  X:[M,HID] f16 row-major, W:[HID,HID] f16 row-major.
// One wave computes a 32(M) x 64(N) strip: 2 A-frags share 4 B-frags per
// K-step (8 WMMAs), halving W traffic.
// MODE 0: plain f32 output [M,HID]                   (final projection)
// MODE 1: Euler-rotate, f16 head-major [B,H,S,DH]    (Q, K)
// MODE 2: f16 TRANSPOSED head-major [B,H,DH,S]       (V)
// ---------------------------------------------------------------------------
template <int MODE>
__global__ __launch_bounds__(256) void gemm_wmma_kernel(
    const _Float16* __restrict__ X, const _Float16* __restrict__ W,
    const float* __restrict__ bias, const float* __restrict__ delta,
    void* __restrict__ out) {
  const int lane    = threadIdx.x & 31;
  const int wid     = blockIdx.x * (blockDim.x >> 5) + (threadIdx.x >> 5);
  const int nStrips = HID_ / 64;               // 8
  const int mt      = wid / nStrips;           // 0..127 (32-row tiles)
  const int ns      = wid % nStrips;

  const _Float16* X0 = X + (size_t)mt * 32 * HID_;
  const _Float16* X1 = X0 + (size_t)16 * HID_;

  v8f acc[2][4];
#pragma unroll
  for (int mi = 0; mi < 2; ++mi)
#pragma unroll
    for (int ng = 0; ng < 4; ++ng) acc[mi][ng] = v8f{0, 0, 0, 0, 0, 0, 0, 0};

  for (int k0 = 0; k0 < HID_; k0 += 32) {
    const v16h a0 = load_A_frag(X0, HID_, k0);
    const v16h a1 = load_A_frag(X1, HID_, k0);
#pragma unroll
    for (int ng = 0; ng < 4; ++ng) {
      const v16h b = load_BT_frag(W, HID_, k0, ns * 64 + ng * 16);
      acc[0][ng] = __builtin_amdgcn_wmma_f32_16x16x32_f16(
          false, a0, false, b, (short)0, acc[0][ng], false, false);
      acc[1][ng] = __builtin_amdgcn_wmma_f32_16x16x32_f16(
          false, a1, false, b, (short)0, acc[1][ng], false, false);
    }
  }

  float dv = 1.0f;
  if constexpr (MODE == 1) dv = delta[0];

  const int half = lane >> 4;
#pragma unroll
  for (int mi = 0; mi < 2; ++mi) {
#pragma unroll
    for (int ng = 0; ng < 4; ++ng) {
#pragma unroll
      for (int g = 0; g < 8; ++g) {
        const int row = mt * 32 + mi * 16 + g + half * 8;  // 0..MROWS-1
        const int col = ns * 64 + ng * 16 + (lane & 15);
        float y = acc[mi][ng][g] + bias[col];
        if constexpr (MODE == 0) {
          ((float*)out)[(size_t)row * HID_ + col] = y;
        } else {
          const int bb = row >> 11;                        // row / S_
          const int s  = row & (S_ - 1);
          const int h  = col >> 6;                         // col / DH_
          const int dh = col & (DH_ - 1);
          if constexpr (MODE == 1) {
            // Euler rotation: pairs (even,odd col) sit in adjacent lanes
            const float other = __shfl_xor(y, 1);
            const bool  even  = ((lane & 1) == 0);
            const float r = even ? y : other;
            const float p = even ? other : y;
            const float lam   = sqrtf(r * r + p * p);
            const int   pid   = dh >> 1;                   // 0..31
            const float freq =
                __expf(-(2.0f * (float)pid / (float)DH_) * LN10000);
            const float theta = atan2f(p, r) * dv + (float)s * freq;
            const float val = even ? lam * __cosf(theta) : lam * __sinf(theta);
            ((_Float16*)out)[(((size_t)(bb * HEADS_ + h)) * S_ + s) * DH_ + dh] =
                (_Float16)val;
          } else {  // MODE 2: V transposed head-major [B,H,DH,S]
            ((_Float16*)out)[(((size_t)(bb * HEADS_ + h)) * DH_ + dh) * S_ + s] =
                (_Float16)y;
          }
        }
      }
    }
  }
}

// ---------------------------------------------------------------------------
// Flash attention (causal). One wave processes TWO Q tiles (qt, QT_MAX-qt) ->
// uniform load. Per 64-key step: 8 QK^T WMMAs + one DPP max butterfly +
// 8 P.V WMMAs + 2 row-sum WMMAs (P x ones: every lane gets its row sum, so
// the final normalization needs no cross-lane reduction at all).
// P goes C-layout -> A-layout via a 2KB/wave LDS buffer (b128 reads).
// V is pre-transposed [B,H,DH,S] so its fragment is 2x b128 per load.
// ---------------------------------------------------------------------------
__global__ __launch_bounds__(256) void attn_wmma_kernel(
    const _Float16* __restrict__ Qh, const _Float16* __restrict__ Kh,
    const _Float16* __restrict__ Vt, _Float16* __restrict__ hidden) {
  alignas(16) __shared__ _Float16 Pbuf[8][16 * 64];  // per-wave 16x64 f16

  const int lane  = threadIdx.x & 31;
  const int wslot = threadIdx.x >> 5;
  const int wid   = blockIdx.x * 8 + wslot;
  const int qtPW  = (QT_MAX + 1) / 2;          // 64 wave-slots per (b,h)
  const int bh    = wid / qtPW;                // 0..B_*HEADS_-1
  const int qsel  = wid % qtPW;
  const int half  = lane >> 4;

  const _Float16* Qb = Qh + (size_t)bh * S_ * DH_;
  const _Float16* Kb = Kh + (size_t)bh * S_ * DH_;
  const _Float16* Vb = Vt + (size_t)bh * DH_ * S_;   // [DH][S]
  const int bb = bh / HEADS_;
  const int h  = bh % HEADS_;
  const float sc = 0.125f;                     // 1/sqrt(DH)

  v16h ones;
#pragma unroll
  for (int i = 0; i < 16; ++i) ones[i] = (_Float16)1.0f;

  const int qts[2] = {qsel, QT_MAX - qsel};

  for (int t = 0; t < 2; ++t) {
    const int qt = qts[t];

    // Q tile fragments (16 x 64 = two 16x32 A-frags), kept in registers
    const v16h aq0 = load_A_frag(Qb + (size_t)qt * 16 * DH_, DH_, 0);
    const v16h aq1 = load_A_frag(Qb + (size_t)qt * 16 * DH_, DH_, 32);

    v8f O[4], Osum;
#pragma unroll
    for (int ng = 0; ng < 4; ++ng) O[ng] = v8f{0, 0, 0, 0, 0, 0, 0, 0};
    Osum = v8f{0, 0, 0, 0, 0, 0, 0, 0};
    float rmax[8];
#pragma unroll
    for (int g = 0; g < 8; ++g) rmax[g] = -__builtin_inff();

    for (int j = 0; j <= qt; j += 4) {         // four 16-key tiles per step
      // ---- QK^T: 4 score tiles (j..j+3; max tile index 127, in-bounds) ----
      v8f s[4];
#pragma unroll
      for (int ti = 0; ti < 4; ++ti) {
        const _Float16* Kt = Kb + (size_t)(j + ti) * 16 * DH_;
        v16h bk = load_BT_frag(Kt, DH_, 0, 0);
        s[ti] = __builtin_amdgcn_wmma_f32_16x16x32_f16(
            false, aq0, false, bk, (short)0, v8f{0, 0, 0, 0, 0, 0, 0, 0},
            false, false);
        bk = load_BT_frag(Kt, DH_, 32, 0);
        s[ti] = __builtin_amdgcn_wmma_f32_16x16x32_f16(
            false, aq1, false, bk, (short)0, s[ti], false, false);
      }

#pragma unroll
      for (int ti = 0; ti < 4; ++ti)
#pragma unroll
        for (int g = 0; g < 8; ++g) s[ti][g] *= sc;

      if (j + 3 >= qt) {                       // only tiles at/past diagonal
#pragma unroll
        for (int g = 0; g < 8; ++g) {
          const int qpos = qt * 16 + g + half * 8;
#pragma unroll
          for (int ti = 0; ti < 4; ++ti) {
            const int kpos = (j + ti) * 16 + (lane & 15);
            if (kpos > qpos) s[ti][g] = -__builtin_inff();
          }
        }
      }

      // ---- online softmax over 64 keys (max only; sums via WMMA) ----
#pragma unroll
      for (int g = 0; g < 8; ++g) {
        float m = fmaxf(fmaxf(s[0][g], s[1][g]), fmaxf(s[2][g], s[3][g]));
        m = rowmax16(m);
        const float nm   = fmaxf(rmax[g], m);
        const float corr = __expf(rmax[g] - nm);
        rmax[g] = nm;
        Osum[g] *= corr;
#pragma unroll
        for (int ng = 0; ng < 4; ++ng) O[ng][g] *= corr;
        const int prow = g + half * 8;
#pragma unroll
        for (int ti = 0; ti < 4; ++ti) {
          const float p = __expf(s[ti][g] - nm);
          Pbuf[wslot][prow * 64 + ti * 16 + (lane & 15)] = (_Float16)p;
        }
      }

      asm volatile("s_wait_dscnt 0" ::: "memory");  // stores visible to reads

      // ---- P.V over 64 keys + row-sum accumulation ----
      const v16h pa0 = load_A_frag(&Pbuf[wslot][0], 64, 0);
      const v16h pa1 = load_A_frag(&Pbuf[wslot][0], 64, 32);
#pragma unroll
      for (int ng = 0; ng < 4; ++ng) {
        v16h bv = load_BT_frag(Vb, S_, j * 16, ng * 16);
        O[ng] = __builtin_amdgcn_wmma_f32_16x16x32_f16(
            false, pa0, false, bv, (short)0, O[ng], false, false);
        bv = load_BT_frag(Vb, S_, j * 16 + 32, ng * 16);
        O[ng] = __builtin_amdgcn_wmma_f32_16x16x32_f16(
            false, pa1, false, bv, (short)0, O[ng], false, false);
      }
      Osum = __builtin_amdgcn_wmma_f32_16x16x32_f16(
          false, pa0, false, ones, (short)0, Osum, false, false);
      Osum = __builtin_amdgcn_wmma_f32_16x16x32_f16(
          false, pa1, false, ones, (short)0, Osum, false, false);
    }

    // ---- normalize (every lane holds its row's sum in Osum[g]) ----
#pragma unroll
    for (int g = 0; g < 8; ++g) {
      const float inv  = 1.0f / Osum[g];
      const int   srow = qt * 16 + g + half * 8;
#pragma unroll
      for (int ng = 0; ng < 4; ++ng) {
        const int dcol = h * DH_ + ng * 16 + (lane & 15);
        hidden[((size_t)bb * S_ + srow) * HID_ + dcol] =
            (_Float16)(O[ng][g] * inv);
      }
    }
  }
}

// ---------------------------------------------------------------------------
// Host-side launch
// ---------------------------------------------------------------------------
extern "C" void kernel_launch(void* const* d_in, const int* in_sizes, int n_in,
                              void* d_out, int out_size, void* d_ws, size_t ws_size,
                              hipStream_t stream) {
  const float* q       = (const float*)d_in[0];
  const float* k       = (const float*)d_in[1];
  const float* v       = (const float*)d_in[2];
  const float* Wq      = (const float*)d_in[3];
  const float* bq      = (const float*)d_in[4];
  const float* Wk      = (const float*)d_in[5];
  const float* bk      = (const float*)d_in[6];
  const float* Wv      = (const float*)d_in[7];
  const float* bv      = (const float*)d_in[8];
  const float* Wo      = (const float*)d_in[9];
  const float* bo      = (const float*)d_in[10];
  const float* delta_q = (const float*)d_in[11];
  const float* delta_k = (const float*)d_in[12];

  // Workspace carve-up (f16), 256B aligned
  char*  ws  = (char*)d_ws;
  size_t off = 0;
  auto carve = [&](size_t elems) {
    void* p = ws + off;
    off += (elems * sizeof(_Float16) + 255) & ~(size_t)255;
    return (_Float16*)p;
  };
  const size_t nAct = (size_t)MROWS * HID_;   // 2,097,152
  const size_t nW   = (size_t)HID_ * HID_;    //   262,144
  _Float16* q16   = carve(nAct);
  _Float16* k16   = carve(nAct);
  _Float16* v16   = carve(nAct);
  _Float16* Wq16  = carve(nW);
  _Float16* Wk16  = carve(nW);
  _Float16* Wv16  = carve(nW);
  _Float16* Wo16  = carve(nW);
  _Float16* Qhm   = carve(nAct);              // [B,H,S,DH] rotated f16
  _Float16* Khm   = carve(nAct);              // [B,H,S,DH] rotated f16
  _Float16* Vtm   = carve(nAct);              // [B,H,DH,S] f16 (transposed)
  _Float16* hid16 = carve(nAct);

  const dim3 blk(256);
  const int cgA = (int)((nAct + 255) / 256);
  const int cgW = (int)((nW + 255) / 256);
  f32_to_f16_kernel<<<cgA, blk, 0, stream>>>(q,  q16,  (int)nAct);
  f32_to_f16_kernel<<<cgA, blk, 0, stream>>>(k,  k16,  (int)nAct);
  f32_to_f16_kernel<<<cgA, blk, 0, stream>>>(v,  v16,  (int)nAct);
  f32_to_f16_kernel<<<cgW, blk, 0, stream>>>(Wq, Wq16, (int)nW);
  f32_to_f16_kernel<<<cgW, blk, 0, stream>>>(Wk, Wk16, (int)nW);
  f32_to_f16_kernel<<<cgW, blk, 0, stream>>>(Wv, Wv16, (int)nW);
  f32_to_f16_kernel<<<cgW, blk, 0, stream>>>(Wo, Wo16, (int)nW);

  // 1024 wave-tiles per GEMM (32x64 strips), 8 waves/block -> 128 blocks
  const int gemmBlocks = (MROWS / 32) * (HID_ / 64) / 8;
  gemm_wmma_kernel<1><<<gemmBlocks, blk, 0, stream>>>(q16, Wq16, bq, delta_q, Qhm);
  gemm_wmma_kernel<1><<<gemmBlocks, blk, 0, stream>>>(k16, Wk16, bk, delta_k, Khm);
  gemm_wmma_kernel<2><<<gemmBlocks, blk, 0, stream>>>(v16, Wv16, bv, nullptr, Vtm);

  // 1024 balanced waves (two Q tiles each), 8 waves/block -> 128 blocks
  const int attnBlocks = (B_ * HEADS_ * ((QT_MAX + 1) / 2)) / 8;
  attn_wmma_kernel<<<attnBlocks, blk, 0, stream>>>(Qhm, Khm, Vtm, hid16);

  gemm_wmma_kernel<0><<<gemmBlocks, blk, 0, stream>>>(hid16, Wo16, bo, nullptr, d_out);
}
